// Proj6_mask_38800734552549
// MI455X (gfx1250) — matrix operand order repaired and checked by reading, not compiled, not run on hardware
//
#include <hip/hip_runtime.h>

// ---------------------------------------------------------------------------
// Fused masked 2-layer MLP for MI455X (gfx1250, wave32, bf16 WMMA + async LDS)
//   out = mask ? (relu(x @ W1 + b1) @ W2 + b2) : 0      B=131072, C=E=1024
//
// Pass 0 (tiny): W1,W2 f32 -> bf16, transposed to Wt[n][k] in d_ws (4 MB).
// Pass 1 (fused): per block 64 rows; X and H tiles live in LDS (CDNA5 320KB
// WGP LDS); W panels streamed with global_load_async_to_lds_b128 into a
// 3-deep ring (one barrier per K-step). Each wave: 2 N-tiles x 4 M-tiles
// accumulators -> 8 WMMAs per 12 ds_load_b128 per step.
// ---------------------------------------------------------------------------

typedef __attribute__((ext_vector_type(16))) __bf16 v16bf;
typedef __attribute__((ext_vector_type(8)))  float  v8f;

#define B_ROWS   131072
#define CDIM     1024
#define EDIM     1024
#define M_TILE   64          // rows per block (4 x 16 M-tiles)
#define N_CHUNK  256         // columns per chunk (8 waves x 32)
#define NC_ITERS (EDIM / N_CHUNK)   // 4
#define K_STEP   32          // one WMMA depth per staged panel
#define K_ITERS  (CDIM / K_STEP)    // 32
#define NBUF     3

#define XS_STRIDE 1032       // bf16 elems: 1024 + 8 pad (16B-aligned groups)
#define WB_STRIDE 40         // bf16 elems per staged column: 32 + 8 pad
#define XS_ELEMS  (M_TILE * XS_STRIDE)      // 66048
#define HS_ELEMS  (M_TILE * XS_STRIDE)      // 66048
#define WBUF_ELEMS (N_CHUNK * WB_STRIDE)    // 10240 per ring slot
#define SMEM_BYTES ((XS_ELEMS + HS_ELEMS + NBUF * WBUF_ELEMS) * 2)  // 325632
static_assert(SMEM_BYTES <= 320 * 1024, "over WGP LDS budget");

__device__ __forceinline__ __bf16 f2bf(float f) { return (__bf16)f; }

// v16bf from two 16B-aligned LDS locations (v16bf wants 32B alignment).
__device__ __forceinline__ v16bf load_frag(const __bf16* p0, const __bf16* p1) {
    union { uint4 u[2]; v16bf v; } t;
    t.u[0] = *(const uint4*)p0;
    t.u[1] = *(const uint4*)p1;
    return t.v;
}

// CDNA5 async global->LDS copy, 16B per lane, tracked by ASYNCcnt.
__device__ __forceinline__ void async_copy_b128(unsigned lds_off, const void* g) {
    asm volatile("global_load_async_to_lds_b128 %0, %1, off"
                 :: "v"(lds_off), "v"(g) : "memory");
}
#define WAIT_ASYNC_LE4() asm volatile("s_wait_asynccnt 0x4" ::: "memory")
#define WAIT_ASYNC_0()   asm volatile("s_wait_asynccnt 0x0" ::: "memory")

// Issue the 16KB panel Wt[ncol0..+255][kk..+31] -> LDS ring slot (4 copies/lane).
__device__ __forceinline__ void stage_w_async(const __bf16* __restrict__ Wt_g,
                                              unsigned slot_off, int ncol0, int kk,
                                              int tid) {
#pragma unroll
    for (int i = 0; i < 4; ++i) {
        int c   = tid + (i << 8);        // chunk id 0..1023
        int col = c >> 2;                // 4 x 16B chunks per column
        int k8  = (c & 3) << 3;          // bf16 k-offset of chunk
        unsigned lds = slot_off + (unsigned)(col * WB_STRIDE + k8) * 2u;
        const void* g = (const void*)(Wt_g + ((size_t)(ncol0 + col) << 10) + kk + k8);
        async_copy_b128(lds, g);
    }
}

// One GEMM layer over this block's 64-row tile.
template <bool TO_LDS>
__device__ __forceinline__ void mlp_layer(
    const __bf16* __restrict__ A,        // LDS activations [64][XS_STRIDE] bf16
    const __bf16* __restrict__ wring,    // LDS ring base (NBUF slots)
    unsigned wring_off,                  // raw LDS byte offset of ring base
    const __bf16* __restrict__ Wt_g,     // global bf16 W, transposed [n][k]
    const float* __restrict__ bias_g,
    __bf16* __restrict__ Hout, float* __restrict__ gout,
    unsigned mbits, int m0, int wave, int lane, int tid)
{
    const int nloc  = lane & 15;
    const int khalf = lane >> 4;
    const int kbase = khalf ? 8 : 0;     // A-fragment K offset (ISA 7.12.2)
    const int rbase = khalf ? 8 : 0;     // C-fragment M offset

    for (int nc = 0; nc < NC_ITERS; ++nc) {
        const int ncol0 = nc * N_CHUNK;

        v8f acc[2][4] = {};              // [n-tile][m-tile], 64 VGPRs

        __syncthreads();   // ring slot 0 free of prior readers
        stage_w_async(Wt_g, wring_off, ncol0, 0, tid);           // batch 0 -> slot 0

        for (int it = 0; it < K_ITERS; ++it) {
            const int kk  = it * K_STEP;
            const int cur = it % NBUF;
            if (it + 1 < K_ITERS) {      // lookahead batch -> next slot
                stage_w_async(Wt_g, wring_off + (unsigned)(((it + 1) % NBUF) * WBUF_ELEMS * 2),
                              ncol0, kk + K_STEP, tid);
                WAIT_ASYNC_LE4();        // retire previous batch (in-order)
            } else {
                WAIT_ASYNC_0();
            }
            __syncthreads();             // everyone's current panel landed

            // B fragments for this wave's two 16-col N-tiles
            const __bf16* wb0 = wring + cur * WBUF_ELEMS
                              + ((wave << 5) + nloc) * WB_STRIDE + khalf * 8;
            const __bf16* wb1 = wb0 + 16 * WB_STRIDE;
            v16bf b0 = load_frag(wb0, wb0 + 16);
            v16bf b1 = load_frag(wb1, wb1 + 16);

            const __bf16* ar = &A[nloc * XS_STRIDE + kk + kbase];
#pragma unroll
            for (int mt = 0; mt < 4; ++mt) {
                const __bf16* ap = ar + mt * 16 * XS_STRIDE;
                v16bf a = load_frag(ap, ap + 16);   // A reused by both N-tiles
                acc[0][mt] = __builtin_amdgcn_wmma_f32_16x16x32_bf16(
                    false, a, false, b0, (short)0, acc[0][mt], false, false);
                acc[1][mt] = __builtin_amdgcn_wmma_f32_16x16x32_bf16(
                    false, a, false, b1, (short)0, acc[1][mt], false, false);
            }
        }

#pragma unroll
        for (int nt = 0; nt < 2; ++nt) {
            const int n = ncol0 + (wave << 5) + (nt << 4) + nloc;
            const float bias = bias_g[n];
            if constexpr (TO_LDS) {      // h = relu(acc + b1) -> LDS bf16
#pragma unroll
                for (int mt = 0; mt < 4; ++mt)
#pragma unroll
                    for (int r = 0; r < 8; ++r) {
                        float v = acc[nt][mt][r] + bias;
                        v = v > 0.0f ? v : 0.0f;
                        Hout[(mt * 16 + rbase + r) * XS_STRIDE + n] = f2bf(v);
                    }
            } else {                     // out = maskbit ? acc + b2 : 0
#pragma unroll
                for (int mt = 0; mt < 4; ++mt)
#pragma unroll
                    for (int r = 0; r < 8; ++r) {
                        int row = m0 + mt * 16 + rbase + r;
                        float v = acc[nt][mt][r] + bias;
                        gout[(size_t)row * EDIM + n] =
                            (mbits >> (mt * 8 + r)) & 1u ? v : 0.0f;
                    }
            }
        }
    }
}

extern "C" __global__ void __launch_bounds__(256, 1)
fused_mlp_kernel(const float* __restrict__ x,
                 const unsigned char* __restrict__ mask,
                 const __bf16* __restrict__ Wt1,
                 const float* __restrict__ b1,
                 const __bf16* __restrict__ Wt2,
                 const float* __restrict__ b2,
                 float* __restrict__ out)
{
    extern __shared__ __bf16 smem[];
    __bf16* Xs    = smem;
    __bf16* Hs    = smem + XS_ELEMS;
    __bf16* wring = smem + XS_ELEMS + HS_ELEMS;
    const unsigned wring_off = (unsigned)(uintptr_t)(void*)wring;  // LDS byte offset

    const int tid  = threadIdx.x;
    const int wave = tid >> 5;
    const int lane = tid & 31;
    const int m0   = blockIdx.x * M_TILE;
    const int rbase = (lane & 16) ? 8 : 0;

    // pack this lane's 32 output-row mask bits once
    unsigned mbits = 0;
#pragma unroll
    for (int mt = 0; mt < 4; ++mt)
#pragma unroll
        for (int r = 0; r < 8; ++r)
            mbits |= (mask[m0 + mt * 16 + rbase + r] ? 1u : 0u) << (mt * 8 + r);

    // stage X tile: 64x1024 f32 -> bf16 LDS (one-time; float4 in, b64 out)
    {
        const float4* xg = (const float4*)(x + (size_t)m0 * CDIM);
        for (int f = tid; f < (M_TILE * CDIM / 4); f += 256) {
            int row = f >> 8;
            int c4  = (f & 255) << 2;
            float4 v = xg[f];
            union { __bf16 h[4]; uint2 u; } p;
            p.h[0] = f2bf(v.x); p.h[1] = f2bf(v.y);
            p.h[2] = f2bf(v.z); p.h[3] = f2bf(v.w);
            *(uint2*)&Xs[row * XS_STRIDE + c4] = p.u;   // 8B-aligned
        }
    }
    __syncthreads();

    // layer 1: Hs = relu(Xs @ W1 + b1)
    mlp_layer<true >(Xs, wring, wring_off, Wt1, b1, Hs, out, mbits, m0, wave, lane, tid);
    __syncthreads();
    // layer 2: out = mask ? (Hs @ W2 + b2) : 0
    mlp_layer<false>(Hs, wring, wring_off, Wt2, b2, Hs, out, mbits, m0, wave, lane, tid);
}

// Pre-pass: W[k][n] f32 -> Wt[n][k] bf16 (4 MB total; runs once per call)
extern "C" __global__ void __launch_bounds__(256)
wt_convert_kernel(const float* __restrict__ W, __bf16* __restrict__ Wt)
{
    int k  = blockIdx.x;             // 0..1023
    int n4 = threadIdx.x << 2;       // coalesced read along n
    float4 v = *(const float4*)&W[((size_t)k << 10) + n4];
    Wt[((size_t)(n4 + 0) << 10) + k] = f2bf(v.x);
    Wt[((size_t)(n4 + 1) << 10) + k] = f2bf(v.y);
    Wt[((size_t)(n4 + 2) << 10) + k] = f2bf(v.z);
    Wt[((size_t)(n4 + 3) << 10) + k] = f2bf(v.w);
}

extern "C" void kernel_launch(void* const* d_in, const int* in_sizes, int n_in,
                              void* d_out, int out_size, void* d_ws, size_t ws_size,
                              hipStream_t stream) {
    const float*         x    = (const float*)d_in[0];
    const unsigned char* mask = (const unsigned char*)d_in[1];  // jnp.bool_ = 1B
    const float*         W1   = (const float*)d_in[2];
    const float*         b1   = (const float*)d_in[3];
    const float*         W2   = (const float*)d_in[4];
    const float*         b2   = (const float*)d_in[5];
    float*               out  = (float*)d_out;
    (void)in_sizes; (void)n_in; (void)out_size; (void)ws_size;

    __bf16* Wt1 = (__bf16*)d_ws;                  // 2 MB
    __bf16* Wt2 = Wt1 + (size_t)CDIM * EDIM;      // 2 MB

    wt_convert_kernel<<<dim3(CDIM), 256, 0, stream>>>(W1, Wt1);
    wt_convert_kernel<<<dim3(EDIM), 256, 0, stream>>>(W2, Wt2);

    hipFuncSetAttribute(reinterpret_cast<const void*>(&fused_mlp_kernel),
                        hipFuncAttributeMaxDynamicSharedMemorySize, SMEM_BYTES);
    fused_mlp_kernel<<<dim3(B_ROWS / M_TILE), 256, SMEM_BYTES, stream>>>(
        x, mask, Wt1, b1, Wt2, b2, out);
}
